// MoeDecoderLayer_64029372449356
// MI455X (gfx1250) — compile-verified, matrix operand order checked by reading, not run on hardware
//
#include <hip/hip_runtime.h>
#include <hip/hip_bf16.h>

// ---------------- constants (match reference) ----------------
#define B_  8
#define S_  256
#define D_  1024
#define H_  16
#define HD_ 64
#define E_  4
#define FF_ 4096
#define M_ROWS (B_*S_)            // 2048
#define YN     (M_ROWS*D_)        // 2,097,152
#define EPS_FILL 2.2204460492503131e-16f

typedef __attribute__((ext_vector_type(16))) __bf16 v16bf;
typedef __attribute__((ext_vector_type(8)))  __bf16 v8bf;
typedef __attribute__((ext_vector_type(8)))  float  v8f;
typedef __attribute__((ext_vector_type(4)))  int    v4i;

#if defined(__gfx1250__) && __has_builtin(__builtin_amdgcn_global_load_async_to_lds_b128) && __has_builtin(__builtin_amdgcn_s_wait_asynccnt)
#define HAVE_ASYNC_LDS 1
#else
#define HAVE_ASYNC_LDS 0
#endif

// float -> bf16 round-to-nearest-even
__device__ __forceinline__ __bf16 f2bf(float f) {
    unsigned u = __float_as_uint(f);
    unsigned r = u + 0x7fffu + ((u >> 16) & 1u);
    unsigned short h = (unsigned short)(r >> 16);
    union { unsigned short s; __bf16 b; } cv; cv.s = h; return cv.b;
}

// Load a 16-bf16 WMMA operand fragment: elements {p[0..7], p[16..23]}
// (with p pre-offset by +8 for the upper half-wave, giving {8..15,24..31}).
__device__ __forceinline__ v16bf ld16g(const __bf16* p) {
    v8bf a = *(const v8bf*)p;
    v8bf b = *(const v8bf*)(p + 16);
    return __builtin_shufflevector(a, b, 0,1,2,3,4,5,6,7,8,9,10,11,12,13,14,15);
}

// ---------------- gating ----------------
__global__ void sum_rows_kernel(const float* __restrict__ x, float* __restrict__ xs) {
    int t = blockIdx.x * 256 + threadIdx.x;
    if (t >= B_ * D_) return;
    int b = t >> 10, d = t & 1023;
    float s = 0.f;
    for (int i = 0; i < S_; ++i) s += x[((size_t)(b * S_ + i)) * D_ + d];
    xs[t] = s;
}

__global__ __launch_bounds__(32)
void gate_kernel(const float* __restrict__ xs, const float* __restrict__ wg,
                 const float* __restrict__ wn, const float* __restrict__ noise,
                 float* __restrict__ gates, float* __restrict__ loss_out) {
    __shared__ float s_clean[32], s_std[32], s_noisy[32], s_g[32], s_prob[32];
    __shared__ float s_thrin[8], s_throut[8];
    int t = threadIdx.x;
    int b = t >> 2, e = t & 3;
    float c = 0.f, rn = 0.f;
    for (int d = 0; d < D_; ++d) {
        float xv = xs[b * D_ + d];
        c  += xv * wg[d * E_ + e];
        rn += xv * wn[d * E_ + e];
    }
    float sp = (rn > 20.f) ? rn : log1pf(__expf(rn));
    float sd = sp + 0.01f;
    float ny = c + noise[b * E_ + e] * sd;
    s_clean[t] = c; s_std[t] = sd; s_noisy[t] = ny; s_g[t] = 0.f;
    __syncthreads();
    if (t < 8) {
        int bb = t;
        float vv[4] = { s_noisy[bb*4+0], s_noisy[bb*4+1], s_noisy[bb*4+2], s_noisy[bb*4+3] };
        int i0 = 0; for (int i = 1; i < 4; ++i) if (vv[i] > vv[i0]) i0 = i;
        int i1 = -1; for (int i = 0; i < 4; ++i) { if (i == i0) continue; if (i1 < 0 || vv[i] > vv[i1]) i1 = i; }
        int i2 = -1; for (int i = 0; i < 4; ++i) { if (i == i0 || i == i1) continue; if (i2 < 0 || vv[i] > vv[i2]) i2 = i; }
        float l0 = vv[i0], l1 = vv[i1], l2 = vv[i2];
        float e1 = __expf(l1 - l0); float ssum = 1.f + e1;
        s_g[bb*4 + i0] = 1.f / ssum;
        s_g[bb*4 + i1] = e1 / ssum;
        s_thrin[bb] = l2;   // (K+1)-th largest
        s_throut[bb] = l1;  // K-th largest
    }
    __syncthreads();
    {
        float thr_in = s_thrin[b], thr_out = s_throut[b];
        bool isin = s_noisy[t] > thr_in;
        float thr = isin ? thr_in : thr_out;
        float z = (s_clean[t] - thr) / s_std[t];
        s_prob[t] = 0.5f * erfcf(-z * 0.70710678118654752f);
        gates[t] = s_g[t];
    }
    __syncthreads();
    if (t == 0) {
        float imp[4] = {0,0,0,0}, ld[4] = {0,0,0,0};
        for (int bb = 0; bb < 8; ++bb)
            for (int ee = 0; ee < 4; ++ee) { imp[ee] += s_g[bb*4+ee]; ld[ee] += s_prob[bb*4+ee]; }
        auto cv = [](const float* v) {
            float m = (v[0]+v[1]+v[2]+v[3]) * 0.25f;
            float s = 0.f; for (int i = 0; i < 4; ++i) { float d = v[i]-m; s += d*d; }
            float var = s / 3.f;               // ddof=1
            return var / (m*m + 1e-10f);
        };
        loss_out[0] = (cv(imp) + cv(ld)) * 0.01f;
    }
}

// ---------------- conversions / packing ----------------
__global__ void cvt_bf16_kernel(const float* __restrict__ x, __bf16* __restrict__ xb, int n) {
    int i = blockIdx.x * 256 + threadIdx.x;
    if (i < n) xb[i] = f2bf(x[i]);
}

// Pack row-major f32 W[Kd][N] into the WMMA B-operand lane layout:
// dst[((nt*KT + kt)*32 + lane)*16 + j]; lane<16 holds K {0..7,16..23},
// lane>=16 holds K {8..15,24..31} for column nt*16+lane%16.
__global__ __launch_bounds__(32)
void pack_w_kernel(const float* __restrict__ W, __bf16* __restrict__ dst, int N, int KT) {
    int kt = blockIdx.x % KT;
    int nt = blockIdx.x / KT;
    int lane = threadIdx.x;
    int n = nt * 16 + (lane & 15);
    int kb = kt * 32;
    int hoff = (lane < 16) ? 0 : 8;
    __bf16* d = dst + (((size_t)nt * KT + kt) * 32 + lane) * 16;
#pragma unroll
    for (int j = 0; j < 8; ++j) d[j]     = f2bf(W[(size_t)(kb + hoff + j) * N + n]);
#pragma unroll
    for (int j = 0; j < 8; ++j) d[8 + j] = f2bf(W[(size_t)(kb + 16 + hoff + j) * N + n]);
}

// ---------------- generic bf16 GEMM (C = A @ W [+bias][relu]) ----------------
// A row-major bf16 MxKd, Bp packed, out f32 and/or bf16.
// Double-buffered LDS B staging; async global->LDS copy on CDNA5 when available.
__global__ __launch_bounds__(256)
void gemm_bf16_kernel(const __bf16* __restrict__ A, const __bf16* __restrict__ Bp,
                      const float* __restrict__ bias, float* __restrict__ Cf,
                      __bf16* __restrict__ Cbf, int M, int N, int Kd, int relu) {
    const int KT = Kd >> 5;
    __shared__ __align__(32) __bf16 ldsB[2][4 * 512];  // 2 x (4 n-tiles x 32 lanes x 16 bf16) = 8KB
    int tid = threadIdx.x;
    int wave = tid >> 5, lane = tid & 31;
    int n0 = blockIdx.x * 64;
    int m0 = (blockIdx.y * 8 + wave) * 16;
    int hf = lane >> 4, l = lane & 15;
    int ntl = tid >> 6;      // 0..3 : which n-tile this thread stages
    int s4  = tid & 63;      // 64 threads x 8 bf16 = 512 bf16 per tile
    const __bf16* arow = A + (size_t)(m0 + l) * Kd + hf * 8;

    auto stage = [&](int kt, int buf) {
        const __bf16* src = Bp + ((size_t)(n0 / 16 + ntl) * KT + kt) * 512 + s4 * 8;
        __bf16* dst = &ldsB[buf][ntl * 512 + s4 * 8];
#if HAVE_ASYNC_LDS
        __builtin_amdgcn_global_load_async_to_lds_b128((v4i*)src, (v4i*)dst, 0, 0);
#else
        *(v8bf*)dst = *(const v8bf*)src;
#endif
    };

    v8f acc0 = {}, acc1 = {}, acc2 = {}, acc3 = {};
    stage(0, 0);
#if HAVE_ASYNC_LDS
    __builtin_amdgcn_s_wait_asynccnt(0);
#endif
    __syncthreads();
    for (int kt = 0; kt < KT; ++kt) {
        int cur = kt & 1;
        if (kt + 1 < KT) stage(kt + 1, cur ^ 1);   // prefetch overlaps WMMAs below
        v16bf a  = ld16g(arow + kt * 32);
        v16bf b0 = *(const v16bf*)&ldsB[cur][0 * 512 + lane * 16];
        v16bf b1 = *(const v16bf*)&ldsB[cur][1 * 512 + lane * 16];
        v16bf b2 = *(const v16bf*)&ldsB[cur][2 * 512 + lane * 16];
        v16bf b3 = *(const v16bf*)&ldsB[cur][3 * 512 + lane * 16];
        acc0 = __builtin_amdgcn_wmma_f32_16x16x32_bf16(false, a, false, b0, (short)0, acc0, false, false);
        acc1 = __builtin_amdgcn_wmma_f32_16x16x32_bf16(false, a, false, b1, (short)0, acc1, false, false);
        acc2 = __builtin_amdgcn_wmma_f32_16x16x32_bf16(false, a, false, b2, (short)0, acc2, false, false);
        acc3 = __builtin_amdgcn_wmma_f32_16x16x32_bf16(false, a, false, b3, (short)0, acc3, false, false);
#if HAVE_ASYNC_LDS
        if (kt + 1 < KT) __builtin_amdgcn_s_wait_asynccnt(0);
#endif
        __syncthreads();
    }
    v8f accs[4] = { acc0, acc1, acc2, acc3 };
#pragma unroll
    for (int nt = 0; nt < 4; ++nt) {
        int n = n0 + nt * 16 + l;
        float bi = bias ? bias[n] : 0.f;
#pragma unroll
        for (int r = 0; r < 8; ++r) {
            int m = m0 + r + hf * 8;
            float v = accs[nt][r] + bi;
            if (relu) v = fmaxf(v, 0.f);
            if (Cf)  Cf[(size_t)m * N + n]  = v;
            if (Cbf) Cbf[(size_t)m * N + n] = f2bf(v);
        }
    }
}

// ---------------- fused causal attention (per b,h, 4 q-tiles) ----------------
__global__ __launch_bounds__(32)
void attn_kernel(const __bf16* __restrict__ Qb, const __bf16* __restrict__ Kb,
                 const __bf16* __restrict__ Vb, __bf16* __restrict__ Ob) {
    __shared__ float  sc[16 * 257];                    // scores, stride 257 (bank-safe)
    __shared__ __align__(16) __bf16 Pl[16 * 272];      // probs bf16
    __shared__ __align__(16) __bf16 Vt[64 * 264];      // V transposed: [hd][key]
    int bid = blockIdx.x;
    int qg = bid & 3;
    int h  = (bid >> 2) & (H_ - 1);
    int b  = bid >> 6;
    int lane = threadIdx.x;
    size_t rowb = (size_t)b * S_;
    int hb = h * HD_;
    // stage V^T into LDS: vector global loads, in-register transpose scatter
    for (int i = lane; i < S_ * 8; i += 32) {
        int s = i >> 3, seg = i & 7;
        v8bf v = *(const v8bf*)(Vb + (rowb + s) * D_ + hb + seg * 8);
#pragma unroll
        for (int j = 0; j < 8; ++j) Vt[(seg * 8 + j) * 264 + s] = v[j];
    }
    __syncthreads();
    int hf = lane >> 4, l = lane & 15;
    for (int qi = 0; qi < 4; ++qi) {
        int qt = qg * 4 + qi;
        int m0 = qt * 16;
        // ---- scores = Q K^T * scale, causal mask ----
        for (int kt = 0; kt <= qt; ++kt) {
            v8f s8 = {};
#pragma unroll
            for (int kk = 0; kk < HD_; kk += 32) {
                v16bf a  = ld16g(Qb + (rowb + m0 + l) * D_ + hb + kk + hf * 8);
                v16bf bb = ld16g(Kb + (rowb + kt * 16 + l) * D_ + hb + kk + hf * 8);
                s8 = __builtin_amdgcn_wmma_f32_16x16x32_bf16(false, a, false, bb, (short)0, s8, false, false);
            }
#pragma unroll
            for (int r = 0; r < 8; ++r) {
                int rl = r + hf * 8;
                float v = s8[r] * 0.125f;                 // 1/sqrt(64)
                if (kt == qt && l > rl) v = -1e9f;        // causal
                sc[rl * 257 + kt * 16 + l] = v;
            }
        }
        __syncthreads();
        // ---- softmax per row (lanes 0..15 handle one row each) ----
        int ncol = (qt + 1) * 16;
        int ncol32 = (ncol + 31) & ~31;
        if (lane < 16) {
            int row = lane;
            float mx = -3.4e38f;
            for (int c = 0; c < ncol; ++c) mx = fmaxf(mx, sc[row * 257 + c]);
            float sum = 0.f;
            for (int c = 0; c < ncol; ++c) {
                float p = __expf(sc[row * 257 + c] - mx);
                sum += p; sc[row * 257 + c] = p;
            }
            float inv = 1.f / sum;
            for (int c = 0; c < ncol; ++c) Pl[row * 272 + c] = f2bf(sc[row * 257 + c] * inv);
            for (int c = ncol; c < ncol32; ++c) Pl[row * 272 + c] = f2bf(0.f);
        }
        __syncthreads();
        // ---- O = P @ V ----
        v8f o[4] = {};
        for (int kc = 0; kc < ncol32; kc += 32) {
            v16bf a = ld16g(&Pl[l * 272 + kc + hf * 8]);
#pragma unroll
            for (int nt = 0; nt < 4; ++nt) {
                v16bf bb = ld16g(&Vt[(nt * 16 + l) * 264 + kc + hf * 8]);
                o[nt] = __builtin_amdgcn_wmma_f32_16x16x32_bf16(false, a, false, bb, (short)0, o[nt], false, false);
            }
        }
#pragma unroll
        for (int nt = 0; nt < 4; ++nt)
#pragma unroll
            for (int r = 0; r < 8; ++r) {
                int rl = r + hf * 8;
                Ob[(rowb + m0 + rl) * D_ + hb + nt * 16 + l] = f2bf(o[nt][r]);
            }
        __syncthreads();
    }
}

// ---------------- residual + LayerNorm ----------------
__global__ __launch_bounds__(256)
void ln_kernel(const float* __restrict__ Xa, const float* __restrict__ Xb,
               const float* __restrict__ g, const float* __restrict__ be,
               float* __restrict__ outf, __bf16* __restrict__ outbf) {
    __shared__ float red[256];
    int row = blockIdx.x, t = threadIdx.x;
    float v[4];
    float p = 0.f;
#pragma unroll
    for (int i = 0; i < 4; ++i) {
        int d = t + i * 256;
        float x = Xa[(size_t)row * D_ + d] + Xb[(size_t)row * D_ + d];
        v[i] = x; p += x;
    }
    red[t] = p; __syncthreads();
    for (int s = 128; s > 0; s >>= 1) { if (t < s) red[t] += red[t + s]; __syncthreads(); }
    float mean = red[0] * (1.f / D_);
    __syncthreads();
    p = 0.f;
#pragma unroll
    for (int i = 0; i < 4; ++i) { float d0 = v[i] - mean; p += d0 * d0; }
    red[t] = p; __syncthreads();
    for (int s = 128; s > 0; s >>= 1) { if (t < s) red[t] += red[t + s]; __syncthreads(); }
    float var = red[0] * (1.f / D_);
    float rs = rsqrtf(var + 1e-5f);
#pragma unroll
    for (int i = 0; i < 4; ++i) {
        int d = t + i * 256;
        float y = (v[i] - mean) * rs * g[d] + be[d];
        if (outf)  outf[(size_t)row * D_ + d]  = y;
        if (outbf) outbf[(size_t)row * D_ + d] = f2bf(y);
    }
}

// ---------------- mixing / output ----------------
__global__ void zero_kernel(float* __restrict__ y, int n) {
    int i = blockIdx.x * 256 + threadIdx.x;
    if (i < n) y[i] = 0.f;
}
__global__ void combine_kernel(float* __restrict__ y, const float* __restrict__ src,
                               const float* __restrict__ gates, int e) {
    int i = blockIdx.x * 256 + threadIdx.x;
    if (i >= YN) return;
    int b = i >> 18;  // S_*D_ = 2^18
    y[i] += gates[b * E_ + e] * src[i];
}
__global__ void epsfill_kernel(float* __restrict__ y, int n) {
    int i = blockIdx.x * 256 + threadIdx.x;
    if (i >= n) return;
    float v = y[i];
    y[i] = (v == 0.f) ? EPS_FILL : v;
}

// ---------------- host launch ----------------
extern "C" void kernel_launch(void* const* d_in, const int* in_sizes, int n_in,
                              void* d_out, int out_size, void* d_ws, size_t ws_size,
                              hipStream_t stream) {
    (void)in_sizes; (void)n_in; (void)out_size; (void)ws_size;
    const float* x      = (const float*)d_in[0];
    const float* noise  = (const float*)d_in[2];
    const float* w_gate = (const float*)d_in[3];
    const float* w_noise= (const float*)d_in[4];
    const float* Wq     = (const float*)d_in[5];
    const float* Wk     = (const float*)d_in[6];
    const float* Wv     = (const float*)d_in[7];
    const float* Wo     = (const float*)d_in[8];
    const float* W1     = (const float*)d_in[9];
    const float* b1     = (const float*)d_in[10];
    const float* W2     = (const float*)d_in[11];
    const float* b2     = (const float*)d_in[12];
    const float* g1     = (const float*)d_in[13];
    const float* be1    = (const float*)d_in[14];
    const float* g2     = (const float*)d_in[15];
    const float* be2    = (const float*)d_in[16];
    float* y = (float*)d_out;
    float* loss = y + YN;

    char* base = (char*)d_ws;
    size_t off = 0;
    auto alloc = [&](size_t bytes) -> void* {
        void* p = base + off;
        off = (off + bytes + 255) & ~(size_t)255;
        return p;
    };
    float*  xs      = (float*) alloc((size_t)B_ * D_ * 4);
    float*  gates   = (float*) alloc(256);
    __bf16* x_bf    = (__bf16*)alloc((size_t)YN * 2);
    __bf16* pw      = (__bf16*)alloc((size_t)D_ * FF_ * 2);      // packed weight (reused), 8MB max
    __bf16* q_bf    = (__bf16*)alloc((size_t)YN * 2);
    __bf16* k_bf    = (__bf16*)alloc((size_t)YN * 2);
    __bf16* v_bf    = (__bf16*)alloc((size_t)YN * 2);
    __bf16* attn_bf = (__bf16*)alloc((size_t)YN * 2);
    float*  o_f32   = (float*) alloc((size_t)YN * 4);
    float*  h_f32   = (float*) alloc((size_t)YN * 4);
    __bf16* h_bf    = (__bf16*)alloc((size_t)YN * 2);
    __bf16* ffa_bf  = (__bf16*)alloc((size_t)M_ROWS * FF_ * 2);
    float*  ff_f32  = (float*) alloc((size_t)YN * 4);
    float*  out_f32 = (float*) alloc((size_t)YN * 4);

    // gating
    sum_rows_kernel<<<32, 256, 0, stream>>>(x, xs);
    gate_kernel<<<1, 32, 0, stream>>>(xs, w_gate, w_noise, noise, gates, loss);
    zero_kernel<<<YN / 256, 256, 0, stream>>>(y, YN);
    cvt_bf16_kernel<<<YN / 256, 256, 0, stream>>>(x, x_bf, YN);

    const dim3 gDD(D_ / 64, M_ROWS / 128);    // GEMMs with N=1024
    const dim3 gDF(FF_ / 64, M_ROWS / 128);   // FFN1, N=4096
    const int packDD = (D_ / 16) * (D_ / 32);     // 2048 blocks
    const int packDF = (FF_ / 16) * (D_ / 32);    // 8192
    const int packFD = (D_ / 16) * (FF_ / 32);    // 8192

    for (int e = 0; e < E_; ++e) {
        const size_t wDD = (size_t)e * D_ * D_;
        const size_t wDF = (size_t)e * D_ * FF_;
        // Q,K,V projections
        pack_w_kernel<<<packDD, 32, 0, stream>>>(Wq + wDD, pw, D_, D_ / 32);
        gemm_bf16_kernel<<<gDD, 256, 0, stream>>>(x_bf, pw, nullptr, nullptr, q_bf, M_ROWS, D_, D_, 0);
        pack_w_kernel<<<packDD, 32, 0, stream>>>(Wk + wDD, pw, D_, D_ / 32);
        gemm_bf16_kernel<<<gDD, 256, 0, stream>>>(x_bf, pw, nullptr, nullptr, k_bf, M_ROWS, D_, D_, 0);
        pack_w_kernel<<<packDD, 32, 0, stream>>>(Wv + wDD, pw, D_, D_ / 32);
        gemm_bf16_kernel<<<gDD, 256, 0, stream>>>(x_bf, pw, nullptr, nullptr, v_bf, M_ROWS, D_, D_, 0);
        // attention
        attn_kernel<<<B_ * H_ * 4, 32, 0, stream>>>(q_bf, k_bf, v_bf, attn_bf);
        // output projection
        pack_w_kernel<<<packDD, 32, 0, stream>>>(Wo + wDD, pw, D_, D_ / 32);
        gemm_bf16_kernel<<<gDD, 256, 0, stream>>>(attn_bf, pw, nullptr, o_f32, nullptr, M_ROWS, D_, D_, 0);
        // LN1: h = LN(x + o)
        ln_kernel<<<M_ROWS, 256, 0, stream>>>(x, o_f32, g1 + e * D_, be1 + e * D_, h_f32, h_bf);
        // FFN
        pack_w_kernel<<<packDF, 32, 0, stream>>>(W1 + wDF, pw, FF_, D_ / 32);
        gemm_bf16_kernel<<<gDF, 256, 0, stream>>>(h_bf, pw, b1 + e * FF_, nullptr, ffa_bf, M_ROWS, FF_, D_, 1);
        pack_w_kernel<<<packFD, 32, 0, stream>>>(W2 + wDF, pw, D_, FF_ / 32);
        gemm_bf16_kernel<<<gDD, 256, 0, stream>>>(ffa_bf, pw, b2 + e * D_, ff_f32, nullptr, M_ROWS, D_, FF_, 0);
        // LN2: out = LN(h + ff)
        ln_kernel<<<M_ROWS, 256, 0, stream>>>(h_f32, ff_f32, g2 + e * D_, be2 + e * D_, out_f32, nullptr);
        // y += gate[b,e] * out
        combine_kernel<<<YN / 256, 256, 0, stream>>>(y, out_f32, gates, e);
    }
    epsfill_kernel<<<YN / 256, 256, 0, stream>>>(y, YN);
}
